// TorchSparseEncoder_67138928771451
// MI455X (gfx1250) — matrix operand order repaired and checked by reading, not compiled, not run on hardware
//
#include <hip/hip_runtime.h>

typedef __attribute__((ext_vector_type(16))) _Float16 v16h;
typedef __attribute__((ext_vector_type(8)))  _Float16 v8h;
typedef __attribute__((ext_vector_type(8)))  float    v8f;

// ---------------- workspace layout (bytes, all 256B aligned) ----------------
static constexpr size_t OFF_GRID = 0;                        // f32 [8*64^3]   8 MB
static constexpr size_t OFF_MASK = 8u  << 20;                // f32 [8*64^3]   8 MB
static constexpr size_t OFF_M2   = 16u << 20;                // f32 [8*32^3]   1 MB
static constexpr size_t OFF_M3   = 17u << 20;                // f32 [8*16^3] 128 KB
static constexpr size_t OFF_S    = (17u << 20) + (1u << 17); // f32 [8*128]
static constexpr size_t OFF_CNT  = OFF_S + 4096;             // f32 [8]
static constexpr size_t OFF_W2F  = OFF_CNT + 256;            // f16 frags, 32 KB
static constexpr size_t OFF_W3F  = OFF_W2F + 32768;          // f16 frags, 128 KB

// ---------------- utility ----------------
__global__ void zero_f32(float* __restrict__ p, int n) {
  int i = blockIdx.x * blockDim.x + threadIdx.x;
  for (; i < n; i += blockDim.x * gridDim.x) p[i] = 0.f;
}

__global__ void scatter_kernel(const int* __restrict__ coords, const float* __restrict__ feats,
                               float* __restrict__ grid, float* __restrict__ mask, int n) {
  int i = blockIdx.x * blockDim.x + threadIdx.x;
  if (i >= n) return;
  int b = coords[4*i+0], x = coords[4*i+1], y = coords[4*i+2], z = coords[4*i+3];
  int v = ((b * 64 + x) * 64 + y) * 64 + z;
  atomicAdd(&grid[v], feats[i]);
  mask[v] = 1.f;
}

// Pre-convert W2/W3 into per-wave WMMA B fragments (f16), fragment-ordered:
// B layout (16x16x32 f16): lane half hh holds K = (e>>3)*16 + hh*8 + (e&7) (e = fragment elem 0..15).
__global__ void prep_weights(const float* __restrict__ W2, const float* __restrict__ W3,
                             _Float16* __restrict__ w2f, _Float16* __restrict__ w3f) {
  int tid = blockIdx.x * blockDim.x + threadIdx.x;
  if (tid < 16384) {                       // 4 cotiles * 8 taps * 32 lanes * 16 elems
    int e = tid & 15, lane = (tid >> 4) & 31, tap = (tid >> 9) & 7, ct = tid >> 12;
    int hh = lane >> 4, r = lane & 15;
    int ci = ((e >> 3) << 4) + hh * 8 + (e & 7);
    w2f[tid] = (_Float16)W2[(tap * 32 + ci) * 64 + ct * 16 + r];
  } else if (tid < 16384 + 65536) {        // 8 cotiles * 16 steps * 32 lanes * 16 elems
    int u = tid - 16384;
    int e = u & 15, lane = (u >> 4) & 31, st = (u >> 9) & 15, ct = u >> 13;
    int hh = lane >> 4, r = lane & 15;
    int tap = st >> 1;
    int ci  = ((st & 1) << 5) + ((e >> 3) << 4) + hh * 8 + (e & 7);
    w3f[u] = (_Float16)W3[(tap * 64 + ci) * 128 + ct * 16 + r];
  }
}

// 2x2x2 max-pool of a mask (Rin = 2*Rout).
__global__ void pool_mask_kernel(const float* __restrict__ src, float* __restrict__ dst, int Rout) {
  int o = blockIdx.x * blockDim.x + threadIdx.x;
  if (o >= 8 * Rout * Rout * Rout) return;
  int oz = o % Rout, t = o / Rout, oy = t % Rout, t2 = t / Rout, ox = t2 % Rout, b = t2 / Rout;
  int Rin = Rout * 2;
  float m = 0.f;
  for (int kd = 0; kd < 2; ++kd)
    for (int kh = 0; kh < 2; ++kh)
      for (int kw = 0; kw < 2; ++kw) {
        float x = src[((b * Rin + 2 * ox + kd) * Rin + 2 * oy + kh) * Rin + 2 * oz + kw];
        m = x > m ? x : m;
      }
  dst[o] = m;
}

// Fully fused conv1 -> conv2 -> conv3 -> masked spatial sum.
// Block = one conv3 z-row tile (b, ox3, oy3, oz=0..15), 256 threads = 8 waves.
// Stride-2 2x2x2 convs have zero input reuse across tiles, so h1/h2 live only in LDS:
//   h1 patch: [4][4][64][32] f16 = 64 KB, h2 patch: [2][2][32][64] f16 = 16 KB.
__global__ void fused_encoder(const float* __restrict__ grid, const float* __restrict__ mask,
                              const float* __restrict__ W1,
                              const _Float16* __restrict__ w2f, const _Float16* __restrict__ w3f,
                              const float* __restrict__ m2, const float* __restrict__ m3,
                              float* __restrict__ s) {
  extern __shared__ char smem[];
  _Float16* h1s = (_Float16*)smem;                // 32768 halves, 64 KB
  _Float16* h2s = (_Float16*)(smem + 65536);      // 8192 halves, 16 KB

  const int tile = blockIdx.x;                    // (b*16+ox3)*16+oy3
  const int oy3 = tile & 15;
  const int ox3 = (tile >> 4) & 15;
  const int b   = tile >> 8;

  const int lane = threadIdx.x & 31;
  const int wv   = threadIdx.x >> 5;              // wave 0..7
  const int hh   = lane >> 4;
  const int r    = lane & 15;

  // ---- Stage A: conv1 (3x3x3, 1->32, SAME, relu*mask) into LDS, sparse early-out ----
  for (int k = 0; k < 4; ++k) {
    const int vox = (int)threadIdx.x + k * 256;   // 0..1023 local voxel
    const int lz = vox & 63, ly = (vox >> 6) & 3, lx = vox >> 8;
    const int gx = ox3 * 4 + lx, gy = oy3 * 4 + ly, gz = lz;
    _Float16* outp = h1s + vox * 32;
    if (mask[((b * 64 + gx) * 64 + gy) * 64 + gz] == 0.f) {
      v8h zv = {};
      *(v8h*)(outp + 0)  = zv; *(v8h*)(outp + 8)  = zv;
      *(v8h*)(outp + 16) = zv; *(v8h*)(outp + 24) = zv;
      continue;
    }
    float acc[32];
#pragma unroll
    for (int c = 0; c < 32; ++c) acc[c] = 0.f;
    for (int dx = -1; dx <= 1; ++dx) {
      int xx = gx + dx; if ((unsigned)xx >= 64u) continue;
      for (int dy = -1; dy <= 1; ++dy) {
        int yy = gy + dy; if ((unsigned)yy >= 64u) continue;
        for (int dz = -1; dz <= 1; ++dz) {
          int zz = gz + dz; if ((unsigned)zz >= 64u) continue;
          float g = grid[((b * 64 + xx) * 64 + yy) * 64 + zz];
          if (g == 0.f) continue;                 // grid ~90% empty
          const float* w = W1 + ((dx + 1) * 9 + (dy + 1) * 3 + (dz + 1)) * 32;
#pragma unroll
          for (int c = 0; c < 32; ++c) acc[c] += g * w[c];
        }
      }
    }
#pragma unroll
    for (int q = 0; q < 4; ++q) {
      v8h o;
#pragma unroll
      for (int j = 0; j < 8; ++j) { float a = acc[q * 8 + j]; o[j] = (_Float16)(a > 0.f ? a : 0.f); }
      *(v8h*)(outp + q * 8) = o;
    }
  }
  __syncthreads();

  // ---- Stage B: conv2 WMMA, LDS h1 patch -> LDS h2 patch ----
  // 32 tasks = 8 M-tiles (lx2,ly2,ozh) x 4 co-tiles, 4 per wave.
  for (int task = wv; task < 32; task += 8) {
    const int mt = task >> 2, ct = task & 3;
    const int ozh = mt & 1, ly2 = (mt >> 1) & 1, lx2 = mt >> 2;
    const int oz2 = ozh * 16 + r;                 // this lane's A row (local output z, 0..31)
    v8f C = {};
#pragma unroll
    for (int t = 0; t < 8; ++t) {
      const int kd = t >> 2, kh = (t >> 1) & 1, kw = t & 1;
      const int lx = 2 * lx2 + kd, ly = 2 * ly2 + kh, lz = 2 * oz2 + kw;
      const _Float16* p = h1s + (((lx * 4 + ly) * 64 + lz) * 32) + (hh << 3);
      v8h lo = *(const v8h*)p;
      v8h hi = *(const v8h*)(p + 16);
      v16h A = __builtin_shufflevector(lo, hi, 0,1,2,3,4,5,6,7,8,9,10,11,12,13,14,15);
      v16h B = *(const v16h*)(w2f + (((ct * 8 + t) * 32 + lane) << 4));
      C = __builtin_amdgcn_wmma_f32_16x16x32_f16(false, A, false, B, (short)0, C, false, false);
    }
    const int gox2 = ox3 * 2 + lx2, goy2 = oy3 * 2 + ly2;
#pragma unroll
    for (int i = 0; i < 8; ++i) {                 // D: M = i + 8*hh, N = r
      const int z2 = ozh * 16 + i + (hh << 3);
      float x = C[i];
      x = x > 0.f ? x : 0.f;
      x *= m2[((b * 32 + gox2) * 32 + goy2) * 32 + z2];
      h2s[(((lx2 * 2 + ly2) * 32 + z2) * 64) + ct * 16 + r] = (_Float16)x;
    }
  }
  __syncthreads();

  // ---- Stage C: conv3 WMMA from LDS + fused masked spatial sum -> s[b][co] ----
  {
    const int oz = r;                             // this lane's A row (output z, 0..15)
    v8f C = {};
#pragma unroll
    for (int st = 0; st < 16; ++st) {
      const int t  = st >> 1, ch = (st & 1) << 5;
      const int kd = t >> 2, kh = (t >> 1) & 1, kw = t & 1;
      const int ivl = (kd * 2 + kh) * 32 + 2 * oz + kw;   // local h2 voxel
      const _Float16* p = h2s + (ivl << 6) + ch + (hh << 3);
      v8h lo = *(const v8h*)p;
      v8h hi = *(const v8h*)(p + 16);
      v16h A = __builtin_shufflevector(lo, hi, 0,1,2,3,4,5,6,7,8,9,10,11,12,13,14,15);
      v16h B = *(const v16h*)(w3f + (((wv * 16 + st) * 32 + lane) << 4));
      C = __builtin_amdgcn_wmma_f32_16x16x32_f16(false, A, false, B, (short)0, C, false, false);
    }
    const int vrow = tile * 16;
    float local = 0.f;
#pragma unroll
    for (int i = 0; i < 8; ++i) {
      float x = C[i];
      x = x > 0.f ? x : 0.f;
      x *= m3[vrow + i + (hh << 3)];
      local += x;
    }
    local += __shfl_xor(local, 16);               // combine the two half-wave M groups
    if (hh == 0) atomicAdd(&s[b * 128 + wv * 16 + r], local);
  }
}

__global__ void cnt_kernel(const float* __restrict__ mask3, float* __restrict__ cnt) {
  __shared__ float sm[256];
  int b = blockIdx.x;
  float t = 0.f;
  for (int i = threadIdx.x; i < 4096; i += 256) t += mask3[b * 4096 + i];
  sm[threadIdx.x] = t;
  __syncthreads();
  for (int off = 128; off > 0; off >>= 1) {
    if ((int)threadIdx.x < off) sm[threadIdx.x] += sm[threadIdx.x + off];
    __syncthreads();
  }
  if (threadIdx.x == 0) cnt[b] = sm[0];
}

__global__ void fc_kernel(const float* __restrict__ s, const float* __restrict__ cnt,
                          const float* __restrict__ fc_w, const float* __restrict__ fc_b,
                          float* __restrict__ out) {
  int b = blockIdx.x, j = threadIdx.x;
  float inv = 1.f / fmaxf(cnt[b], 1.f);
  float acc = fc_b[j];
  for (int k = 0; k < 128; ++k) acc += (s[b * 128 + k] * inv) * fc_w[k * 256 + j];
  out[b * 256 + j] = acc;
}

extern "C" void kernel_launch(void* const* d_in, const int* in_sizes, int n_in,
                              void* d_out, int out_size, void* d_ws, size_t ws_size,
                              hipStream_t stream) {
  const int*   coords = (const int*)d_in[0];
  const float* feats  = (const float*)d_in[1];
  const float* W1     = (const float*)d_in[2];
  const float* W2     = (const float*)d_in[3];
  const float* W3     = (const float*)d_in[4];
  const float* fc_w   = (const float*)d_in[5];
  const float* fc_b   = (const float*)d_in[6];
  float* out = (float*)d_out;
  const int N = in_sizes[0] / 4;

  char* ws = (char*)d_ws;
  float*    grid = (float*)(ws + OFF_GRID);
  float*    mask = (float*)(ws + OFF_MASK);
  float*    m2   = (float*)(ws + OFF_M2);
  float*    m3   = (float*)(ws + OFF_M3);
  float*    sbuf = (float*)(ws + OFF_S);
  float*    cnt  = (float*)(ws + OFF_CNT);
  _Float16* w2f  = (_Float16*)(ws + OFF_W2F);
  _Float16* w3f  = (_Float16*)(ws + OFF_W3F);

  zero_f32<<<4096, 256, 0, stream>>>(grid, 2 * 2097152);   // grid + mask (contiguous)
  zero_f32<<<8, 256, 0, stream>>>(sbuf, 1032);             // s (1024) + cnt (8, contiguous)
  prep_weights<<<320, 256, 0, stream>>>(W2, W3, w2f, w3f);
  scatter_kernel<<<(N + 255) / 256, 256, 0, stream>>>(coords, feats, grid, mask, N);
  pool_mask_kernel<<<1024, 256, 0, stream>>>(mask, m2, 32);
  pool_mask_kernel<<<128, 256, 0, stream>>>(m2, m3, 16);
  fused_encoder<<<2048, 256, 81920, stream>>>(grid, mask, W1, w2f, w3f, m2, m3, sbuf);
  cnt_kernel<<<8, 256, 0, stream>>>(m3, cnt);
  fc_kernel<<<8, 256, 0, stream>>>(sbuf, cnt, fc_w, fc_b, out);
}
